// CrossAttention_23897198034990
// MI455X (gfx1250) — matrix operand order repaired and checked by reading, not compiled, not run on hardware
//
#include <hip/hip_runtime.h>
#include <hip/hip_bf16.h>

// ---------------------------------------------------------------------------
// CDNA5 (gfx1250, wave32) fused cross-attention:
//   q = query@Wq+bq ; k,v = kv@Wkv+bkv ; flash-attn with ALiBi ; out = attn@Wo+bo
// All matmuls run on v_wmma_f32_16x16x32_f16 (f16 operands, f32 accum).
// f16 LDS staging uses GLOBAL_LOAD_ASYNC_TO_LDS_B128 (ASYNCcnt) when the
// toolchain exposes the builtin; falls back to VGPR copies otherwise.
// Workspace layout (f16): Q [B,H,Nq,D] | K [B,H,Nk,D] | V^T [B,H,D,Nk] | attnout [B*Nq, E]
// = 4 * 8MB = 32MB.
// ---------------------------------------------------------------------------

typedef _Float16 v16h __attribute__((ext_vector_type(16)));
typedef _Float16 v8h  __attribute__((ext_vector_type(8)));
typedef float    v8f  __attribute__((ext_vector_type(8)));
typedef int      v4i  __attribute__((vector_size(16)));   // matches builtin param

#define EMBED 1024
#define HEADS 16
#define HDIM  64
#define NSEQ  2048

#if __has_builtin(__builtin_amdgcn_global_load_async_to_lds_b128) && \
    __has_builtin(__builtin_amdgcn_s_wait_asynccnt)
#define USE_ASYNC_LDS 1
typedef __attribute__((address_space(1))) v4i* g_v4i_ptr;  // global (AS1)
typedef __attribute__((address_space(3))) v4i* l_v4i_ptr;  // LDS (AS3)
#endif

// Copy 16 bytes global -> LDS. Async path: one GLOBAL_LOAD_ASYNC_TO_LDS_B128
// per lane (no VGPR round trip, tracked by ASYNCcnt).
__device__ __forceinline__ void copy16_g2l(const _Float16* g, _Float16* l) {
#ifdef USE_ASYNC_LDS
  __builtin_amdgcn_global_load_async_to_lds_b128((g_v4i_ptr)g, (l_v4i_ptr)l, 0, 0);
#else
  *(v8h*)l = *(const v8h*)g;
#endif
}

__device__ __forceinline__ void wait_async_copies() {
#ifdef USE_ASYNC_LDS
  __builtin_amdgcn_s_wait_asynccnt(0);
#endif
}

__device__ __forceinline__ v8f wmma_f16(v16h a, v16h b, v8f c) {
  // (neg_a, A, neg_b, B, c_mod, C, reuse_a, reuse_b)
  return __builtin_amdgcn_wmma_f32_16x16x32_f16(false, a, false, b, (short)0, c,
                                                false, false);
}

// A-matrix fragment (16x32 f16): lane holds row (lane&15); halves 0..7 cover
// K = 8*(lane>>4) .. +8, halves 8..15 cover K = 16 + 8*(lane>>4) .. +8.
__device__ __forceinline__ v16h make_afrag(const _Float16* rowPtr, int lane) {
  const int off = (lane >> 4) * 8;
  v8h lo = *(const v8h*)(rowPtr + off);
  v8h hi = *(const v8h*)(rowPtr + 16 + off);
  v16h a;
#pragma unroll
  for (int i = 0; i < 8; ++i) { a[i] = lo[i]; a[i + 8] = hi[i]; }
  return a;
}

// B-matrix fragment (32x16 f16, fed from B^T rows): lane holds column (lane&15);
// halves 0..7 cover K = 8*(lane>>4) .. +8, halves 8..15 cover K = 16 + 8*(lane>>4) .. +8.  colPtr points at B^T[col][0].
__device__ __forceinline__ v16h make_bfrag(const _Float16* colPtr, int lane) {
  const int off = (lane >> 4) * 8;
  v8h lo = *(const v8h*)(colPtr + off);
  v8h hi = *(const v8h*)(colPtr + 16 + off);
  v16h b;
#pragma unroll
  for (int i = 0; i < 8; ++i) { b[i] = lo[i]; b[i + 8] = hi[i]; }
  return b;
}

// ---------------------------------------------------------------------------
// Projection GEMM: out(f16) = X(f32)[4096 x 1024] @ W(f32)[1024 x ldW] + bias
// mode 0: out[((b*H+h)*NSEQ+n)*64+d]   (Q and K, per-head row-major, d contig)
// mode 1: out[((b*H+h)*64+d)*NSEQ+n]   (V transposed, n contig)
// Block: 128 thr (4 waves), 128x128 tile; wave: 64x64 = 4x4 WMMA tiles.
// (f32 sources must pass through VGPRs for the f32->f16 convert, so this
//  kernel keeps the load+cvt+ds_store staging path.)
// ---------------------------------------------------------------------------
__global__ __launch_bounds__(128)
void proj_gemm_kernel(const float* __restrict__ X, const float* __restrict__ W,
                      const float* __restrict__ bias, _Float16* __restrict__ out,
                      int ldW, int mode) {
  __shared__ _Float16 As[128][40];  // [m][k], stride 80B (16B aligned)
  __shared__ _Float16 Bs[128][40];  // [n][k] == W^T tile

  const int t = threadIdx.x;
  const int lane = t & 31;
  const int w = t >> 5;
  const int rowBase = blockIdx.x * 128;
  const int colBase = blockIdx.y * 128;
  const int m0w = (w >> 1) * 64;
  const int n0w = (w & 1) * 64;
  const int mrow = lane & 15;

  const v8f vz = {0.f, 0.f, 0.f, 0.f, 0.f, 0.f, 0.f, 0.f};
  v8f acc[4][4];
#pragma unroll
  for (int mi = 0; mi < 4; ++mi)
#pragma unroll
    for (int ni = 0; ni < 4; ++ni) acc[mi][ni] = vz;

  for (int kb = 0; kb < EMBED; kb += 32) {
    // stage A tile: 128 rows x 32 k, f32 -> f16
    {
      const float* src = X + (size_t)(rowBase + t) * EMBED + kb;
#pragma unroll
      for (int q = 0; q < 8; ++q) {
        float4 f = *(const float4*)(src + 4 * q);
        As[t][4 * q + 0] = (_Float16)f.x;
        As[t][4 * q + 1] = (_Float16)f.y;
        As[t][4 * q + 2] = (_Float16)f.z;
        As[t][4 * q + 3] = (_Float16)f.w;
      }
    }
    // stage B tile transposed: Bs[n][k] from W[k][n]
#pragma unroll
    for (int q = 0; q < 8; ++q) {
      const int f = t + 128 * q;
      const int k = f >> 5;
      const int nc = (f & 31) * 4;
      float4 ww = *(const float4*)(W + (size_t)(kb + k) * ldW + colBase + nc);
      Bs[nc + 0][k] = (_Float16)ww.x;
      Bs[nc + 1][k] = (_Float16)ww.y;
      Bs[nc + 2][k] = (_Float16)ww.z;
      Bs[nc + 3][k] = (_Float16)ww.w;
    }
    if (kb + 32 < EMBED)
      __builtin_prefetch(W + (size_t)(kb + 32) * ldW + colBase, 0, 3);
    __syncthreads();

    v16h afr[4];
#pragma unroll
    for (int mi = 0; mi < 4; ++mi)
      afr[mi] = make_afrag(&As[m0w + mi * 16 + mrow][0], lane);
#pragma unroll
    for (int ni = 0; ni < 4; ++ni) {
      v16h bfr = make_bfrag(&Bs[n0w + ni * 16 + mrow][0], lane);
#pragma unroll
      for (int mi = 0; mi < 4; ++mi)
        acc[mi][ni] = wmma_f16(afr[mi], bfr, acc[mi][ni]);
    }
    __syncthreads();
  }

  // epilogue: C layout M = v + 8*(lane>=16), N = lane&15
  const int mshift = (lane >> 4) * 8;
  const int ncol = lane & 15;
#pragma unroll
  for (int mi = 0; mi < 4; ++mi) {
#pragma unroll
    for (int ni = 0; ni < 4; ++ni) {
#pragma unroll
      for (int v = 0; v < 8; ++v) {
        const int r = rowBase + m0w + mi * 16 + v + mshift;
        const int c = colBase + n0w + ni * 16 + ncol;
        const float val = acc[mi][ni][v] + bias[c];
        const int bb = r >> 11, n = r & (NSEQ - 1);
        const int h = c >> 6, d = c & 63;
        size_t idx;
        if (mode == 0)
          idx = ((size_t)(bb * HEADS + h) * NSEQ + n) * HDIM + d;
        else
          idx = ((size_t)(bb * HEADS + h) * HDIM + d) * NSEQ + n;
        out[idx] = (_Float16)val;
      }
    }
  }
}

// ---------------------------------------------------------------------------
// Flash attention with ALiBi. Block = 128 thr (4 waves) handles one (b,h) and
// 64 query rows; each wave owns 16 rows. K-chunks of 64 staged in LDS via
// async LDS-DMA (f16 data, no conversion needed).
// ---------------------------------------------------------------------------
__global__ __launch_bounds__(128)
void attn_kernel(const _Float16* __restrict__ Q, const _Float16* __restrict__ K,
                 const _Float16* __restrict__ Vt, _Float16* __restrict__ attnout) {
  __shared__ _Float16 Ks[64][72];      // [k_local][d]
  __shared__ _Float16 Vs[64][72];      // [d][k_local]  (V already transposed)
  __shared__ _Float16 Ps[4][16][72];   // per-wave P bounce [q_local][k_local]

  const int t = threadIdx.x;
  const int lane = t & 31;
  const int w = t >> 5;
  const int qt = blockIdx.x & 31;
  const int hb = blockIdx.x >> 5;
  const int h = hb & (HEADS - 1);
  const int b = hb >> 4;
  const int q0w = qt * 64 + w * 16;
  const int mrow = lane & 15;
  const int mshift = (lane >> 4) * 8;

  const _Float16* Qp = Q + (size_t)(b * HEADS + h) * NSEQ * HDIM;
  const _Float16* Kp = K + (size_t)(b * HEADS + h) * NSEQ * HDIM;
  const _Float16* Vp = Vt + (size_t)(b * HEADS + h) * HDIM * NSEQ;

  const float slope = exp2f(-8.0f * (float)(h + 1) / (float)HEADS);
  const float scale = 0.125f;  // 1/sqrt(64)

  // resident Q fragments (16 rows x 64 d = two A-frags)
  const v16h qf0 = make_afrag(Qp + (size_t)(q0w + mrow) * HDIM, lane);
  const v16h qf1 = make_afrag(Qp + (size_t)(q0w + mrow) * HDIM + 32, lane);

  const v8f vz = {0.f, 0.f, 0.f, 0.f, 0.f, 0.f, 0.f, 0.f};
  v8f o[4];
#pragma unroll
  for (int ni = 0; ni < 4; ++ni) o[ni] = vz;
  float m[8], l[8];
#pragma unroll
  for (int v = 0; v < 8; ++v) { m[v] = -1e30f; l[v] = 0.f; }

  for (int kc = 0; kc < NSEQ; kc += 64) {
    // cooperative stage of K and V^T chunks (64x64 f16 each) via async LDS-DMA
    {
      const int rr = t >> 1;
      const int off = (t & 1) * 32;
#pragma unroll
      for (int j = 0; j < 4; ++j) {
        copy16_g2l(Kp + (size_t)(kc + rr) * HDIM + off + 8 * j,
                   &Ks[rr][off + 8 * j]);
        copy16_g2l(Vp + (size_t)rr * NSEQ + kc + off + 8 * j,
                   &Vs[rr][off + 8 * j]);
      }
      wait_async_copies();
    }
    __syncthreads();

    // S = Q K^T * scale + alibi   (4 column subtiles of 16)
    float sv[4][8];
#pragma unroll
    for (int sub = 0; sub < 4; ++sub) {
      v16h kf0 = make_bfrag(&Ks[sub * 16 + mrow][0], lane);
      v16h kf1 = make_bfrag(&Ks[sub * 16 + mrow][32], lane);
      v8f s = vz;
      s = wmma_f16(qf0, kf0, s);
      s = wmma_f16(qf1, kf1, s);
      const int jcol = kc + sub * 16 + (lane & 15);
#pragma unroll
      for (int v = 0; v < 8; ++v) {
        const int iq = q0w + v + mshift;
        sv[sub][v] = s[v] * scale - slope * fabsf((float)(jcol - iq));
      }
    }

    // online softmax: row max over 4 subtiles + 16-lane halves
    float corr[8];
#pragma unroll
    for (int v = 0; v < 8; ++v) {
      float x = fmaxf(fmaxf(sv[0][v], sv[1][v]), fmaxf(sv[2][v], sv[3][v]));
      x = fmaxf(x, __shfl_xor(x, 1, 32));
      x = fmaxf(x, __shfl_xor(x, 2, 32));
      x = fmaxf(x, __shfl_xor(x, 4, 32));
      x = fmaxf(x, __shfl_xor(x, 8, 32));
      const float mnew = fmaxf(m[v], x);
      corr[v] = __expf(m[v] - mnew);
      m[v] = mnew;
    }

    // P = exp(S - m), bounce through LDS into A-fragment layout
#pragma unroll
    for (int sub = 0; sub < 4; ++sub) {
#pragma unroll
      for (int v = 0; v < 8; ++v) {
        const float p = __expf(sv[sub][v] - m[v]);
        sv[sub][v] = p;
        Ps[w][v + mshift][sub * 16 + (lane & 15)] = (_Float16)p;
      }
    }
#pragma unroll
    for (int v = 0; v < 8; ++v) {
      float rs = sv[0][v] + sv[1][v] + sv[2][v] + sv[3][v];
      rs += __shfl_xor(rs, 1, 32);
      rs += __shfl_xor(rs, 2, 32);
      rs += __shfl_xor(rs, 4, 32);
      rs += __shfl_xor(rs, 8, 32);
      l[v] = l[v] * corr[v] + rs;
    }
#pragma unroll
    for (int ni = 0; ni < 4; ++ni)
#pragma unroll
      for (int v = 0; v < 8; ++v) o[ni][v] *= corr[v];

    // O += P @ V   (P: 16x64 -> two A-frags; V^T rows feed B-frags)
    const v16h pf0 = make_afrag(&Ps[w][mrow][0], lane);
    const v16h pf1 = make_afrag(&Ps[w][mrow][32], lane);
#pragma unroll
    for (int ni = 0; ni < 4; ++ni) {
      v16h vf0 = make_bfrag(&Vs[ni * 16 + mrow][0], lane);
      v16h vf1 = make_bfrag(&Vs[ni * 16 + mrow][32], lane);
      o[ni] = wmma_f16(pf0, vf0, o[ni]);
      o[ni] = wmma_f16(pf1, vf1, o[ni]);
    }
    __syncthreads();
  }

  // normalize and write attnout[b*NSEQ + q][h*64 + d] (f16 row-major)
#pragma unroll
  for (int ni = 0; ni < 4; ++ni) {
#pragma unroll
    for (int v = 0; v < 8; ++v) {
      const int M = v + mshift;
      const size_t idx = (size_t)(b * NSEQ + q0w + M) * EMBED + h * HDIM +
                         ni * 16 + (lane & 15);
      attnout[idx] = (_Float16)(o[ni][v] / l[v]);
    }
  }
}

// ---------------------------------------------------------------------------
// Output GEMM: d_out(f32) = attnout(f16)[4096 x 1024] @ Wo(f32) + bo
// A tile is already f16 -> staged with async LDS-DMA.
// ---------------------------------------------------------------------------
__global__ __launch_bounds__(128)
void out_gemm_kernel(const _Float16* __restrict__ A, const float* __restrict__ W,
                     const float* __restrict__ bias, float* __restrict__ out) {
  __shared__ _Float16 As[128][40];
  __shared__ _Float16 Bs[128][40];

  const int t = threadIdx.x;
  const int lane = t & 31;
  const int w = t >> 5;
  const int rowBase = blockIdx.x * 128;
  const int colBase = blockIdx.y * 128;
  const int m0w = (w >> 1) * 64;
  const int n0w = (w & 1) * 64;
  const int mrow = lane & 15;

  const v8f vz = {0.f, 0.f, 0.f, 0.f, 0.f, 0.f, 0.f, 0.f};
  v8f acc[4][4];
#pragma unroll
  for (int mi = 0; mi < 4; ++mi)
#pragma unroll
    for (int ni = 0; ni < 4; ++ni) acc[mi][ni] = vz;

  for (int kb = 0; kb < EMBED; kb += 32) {
    {
      const _Float16* src = A + (size_t)(rowBase + t) * EMBED + kb;
#pragma unroll
      for (int q = 0; q < 4; ++q)
        copy16_g2l(src + 8 * q, &As[t][8 * q]);
    }
#pragma unroll
    for (int q = 0; q < 8; ++q) {
      const int f = t + 128 * q;
      const int k = f >> 5;
      const int nc = (f & 31) * 4;
      float4 ww = *(const float4*)(W + (size_t)(kb + k) * EMBED + colBase + nc);
      Bs[nc + 0][k] = (_Float16)ww.x;
      Bs[nc + 1][k] = (_Float16)ww.y;
      Bs[nc + 2][k] = (_Float16)ww.z;
      Bs[nc + 3][k] = (_Float16)ww.w;
    }
    wait_async_copies();
    __syncthreads();

    v16h afr[4];
#pragma unroll
    for (int mi = 0; mi < 4; ++mi)
      afr[mi] = make_afrag(&As[m0w + mi * 16 + mrow][0], lane);
#pragma unroll
    for (int ni = 0; ni < 4; ++ni) {
      v16h bfr = make_bfrag(&Bs[n0w + ni * 16 + mrow][0], lane);
#pragma unroll
      for (int mi = 0; mi < 4; ++mi)
        acc[mi][ni] = wmma_f16(afr[mi], bfr, acc[mi][ni]);
    }
    __syncthreads();
  }

  const int mshift = (lane >> 4) * 8;
  const int ncol = lane & 15;
#pragma unroll
  for (int mi = 0; mi < 4; ++mi)
#pragma unroll
    for (int ni = 0; ni < 4; ++ni)
#pragma unroll
      for (int v = 0; v < 8; ++v) {
        const int r = rowBase + m0w + mi * 16 + v + mshift;
        const int c = colBase + n0w + ni * 16 + ncol;
        out[(size_t)r * EMBED + c] = acc[mi][ni][v] + bias[c];
      }
}

// ---------------------------------------------------------------------------
extern "C" void kernel_launch(void* const* d_in, const int* in_sizes, int n_in,
                              void* d_out, int out_size, void* d_ws, size_t ws_size,
                              hipStream_t stream) {
  const float* query = (const float*)d_in[0];  // [2,2048,1024]
  const float* kv    = (const float*)d_in[1];  // [2,2048,1024]
  const float* Wq    = (const float*)d_in[2];  // [1024,1024]
  const float* bq    = (const float*)d_in[3];
  const float* Wkv   = (const float*)d_in[4];  // [1024,2048]
  const float* bkv   = (const float*)d_in[5];
  const float* Wo    = (const float*)d_in[6];  // [1024,1024]
  const float* bo    = (const float*)d_in[7];
  float* out = (float*)d_out;                  // [2,2048,1024] f32

  _Float16* ws = (_Float16*)d_ws;
  const size_t SZ = (size_t)2 * HEADS * NSEQ * HDIM;  // 4M halves
  _Float16* Qw = ws;
  _Float16* Kw = ws + SZ;
  _Float16* Vw = ws + 2 * SZ;
  _Float16* Aw = ws + 3 * SZ;

  dim3 gG(32, 8), bG(128);
  // Q = query@Wq+bq  -> per-head [b,h,n,d]
  proj_gemm_kernel<<<gG, bG, 0, stream>>>(query, Wq, bq, Qw, EMBED, 0);
  // K = kv@Wkv[:, :E]+bkv[:E] -> per-head [b,h,n,d]
  proj_gemm_kernel<<<gG, bG, 0, stream>>>(kv, Wkv, bkv, Kw, 2 * EMBED, 0);
  // V = kv@Wkv[:, E:]+bkv[E:] -> transposed per-head [b,h,d,n]
  proj_gemm_kernel<<<gG, bG, 0, stream>>>(kv, Wkv + EMBED, bkv + EMBED, Vw,
                                          2 * EMBED, 1);
  // flash attention with ALiBi
  attn_kernel<<<dim3(2 * HEADS * (NSEQ / 64)), 128, 0, stream>>>(Qw, Kw, Vw, Aw);
  // out = attnout@Wo + bo
  out_gemm_kernel<<<gG, bG, 0, stream>>>(Aw, Wo, bo, out);
}
